// GCN_low_65283502899906
// MI455X (gfx1250) — matrix operand order repaired and checked by reading, not compiled, not run on hardware
//
#include <hip/hip_runtime.h>
#include <cstdint>

typedef __attribute__((ext_vector_type(16))) _Float16 v16h;
typedef __attribute__((ext_vector_type(8)))  _Float16 v8h;
typedef __attribute__((ext_vector_type(8)))  float    v8f;

#define FDIM 256
#define BN_EPS 1e-5f

__device__ __forceinline__ void atomic_add_f32(float* p, float v) {
  unsafeAtomicAdd(p, v);   // native global_atomic_add_f32 (no-return)
}

// ----------------------------------------------------------------------------
// GEMM: C[N,256] = act(A)[N,256] x W[256,256]; f32 in, f16 WMMA, f32 out.
// FUSE_BN: act(x) = relu(x * sb[c] + sb[256+c]) applied during A staging.
// Dual store: every C element is written to both C and C2 (self-loop init).
//
// Block = 256 threads (8 waves). Block tile: 128 rows x 64 cols.
// Wave w computes rows [r0+16w, r0+16w+16) x all 4 column subtiles.
//
// Fragment packing (wave32, 16-bit, CDNA5 ISA 7.12.2):
//  A(16x32): element (m,kk): lane = m + 16*(kk>=16),    half = kk & 15
//    -> an aligned octet of K is 8 contiguous halves in one lane.
//  B(32x16): element (kk,n): lane = n + 16*(kk>=16),    half = kk & 15
//    -> an aligned octet of K at fixed n is 8 contiguous halves in one lane.
//  C(16x16 f32): acc[v] at lane L: row = v + 8*(L>=16), col = L&15
// ----------------------------------------------------------------------------
template <bool FUSE_BN>
__global__ __launch_bounds__(256) void gemm_wmma(const float* __restrict__ A,
                                                 const float* __restrict__ W,
                                                 const float* __restrict__ sb,
                                                 float* __restrict__ C,
                                                 float* __restrict__ C2,
                                                 int N) {
  __shared__ __align__(16) _Float16 sA[8 * 512];    //  8 KB (current K-slice)
  __shared__ __align__(16) _Float16 sB[32 * 512];   // 32 KB (entire 256x64 W block)
  const int t    = threadIdx.x;
  const int lane = t & 31;
  const int mb   = t >> 5;            // wave id == M subtile
  const int r0   = blockIdx.x * 128;
  const int c0   = blockIdx.y * 64;

  // ---- Stage all of B (256 x 64) into LDS in fragment order, f32 -> f16.
  // item = ko*64 + n : for fixed ko, consecutive threads -> consecutive n.
  #pragma unroll
  for (int it = 0; it < 8; ++it) {
    int item = it * 256 + t;
    int n  = item & 63;
    int ko = item >> 6;
    int k0b = ko * 8;
    const float* src = W + k0b * FDIM + c0 + n;
    v8f f;
    #pragma unroll
    for (int j = 0; j < 8; ++j) f[j] = src[j * FDIM];
    v8h h = __builtin_convertvector(f, v8h);
    int kk0 = k0b & 31;
    int nb = n >> 4, nn = n & 15;
    int bl  = nn + ((kk0 & 16) ? 16 : 0);
    int bh0 = kk0 & 8;
    *(v8h*)(sB + ((k0b >> 5) * 4 + nb) * 512 + bl * 16 + bh0) = h;
  }

  v8f acc0 = {}, acc1 = {}, acc2 = {}, acc3 = {};

  for (int kbi = 0; kbi < 8; ++kbi) {
    __syncthreads();
    // ---- Stage A K-slice: 128 rows x 32 k. item = rowRel*4 + octet, 2/thread.
    const int k0 = kbi * 32;
    #pragma unroll
    for (int it = 0; it < 2; ++it) {
      int item   = it * 256 + t;
      int o      = item & 3;           // K-octet within the 32-wide slice
      int rowRel = item >> 2;
      int row    = r0 + rowRel;
      int rc     = min(row, N - 1);    // branchless clamp; zero-select below
      const float4* src = (const float4*)(A + rc * FDIM + k0 + o * 8);
      float4 x0 = src[0];
      float4 x1 = src[1];
      if (row >= N) {
        x0 = make_float4(0.f, 0.f, 0.f, 0.f);
        x1 = x0;
      }
      v8f f = {x0.x, x0.y, x0.z, x0.w, x1.x, x1.y, x1.z, x1.w};
      if (FUSE_BN) {
        // act(x) = relu(x * scale[c] + bias[c]), c = k0 + o*8 + j (32B aligned)
        v8f sc = *(const v8f*)(sb + k0 + o * 8);
        v8f bs = *(const v8f*)(sb + FDIM + k0 + o * 8);
        #pragma unroll
        for (int j = 0; j < 8; ++j) f[j] = fmaxf(fmaf(f[j], sc[j], bs[j]), 0.0f);
      }
      v8h h = __builtin_convertvector(f, v8h);
      int m = rowRel & 15, mbs = rowRel >> 4;
      int al  = m + ((o & 2) ? 16 : 0);
      int ah0 = (o & 1) ? 8 : 0;
      *(v8h*)(sA + mbs * 512 + al * 16 + ah0) = h;
    }
    __syncthreads();

    v16h a  = *(const v16h*)(sA + mb * 512 + lane * 16);
    v16h b0 = *(const v16h*)(sB + (kbi * 4 + 0) * 512 + lane * 16);
    v16h b1 = *(const v16h*)(sB + (kbi * 4 + 1) * 512 + lane * 16);
    v16h b2 = *(const v16h*)(sB + (kbi * 4 + 2) * 512 + lane * 16);
    v16h b3 = *(const v16h*)(sB + (kbi * 4 + 3) * 512 + lane * 16);
    acc0 = __builtin_amdgcn_wmma_f32_16x16x32_f16(false, a, false, b0, (short)0, acc0, false, false);
    acc1 = __builtin_amdgcn_wmma_f32_16x16x32_f16(false, a, false, b1, (short)0, acc1, false, false);
    acc2 = __builtin_amdgcn_wmma_f32_16x16x32_f16(false, a, false, b2, (short)0, acc2, false, false);
    acc3 = __builtin_amdgcn_wmma_f32_16x16x32_f16(false, a, false, b3, (short)0, acc3, false, false);
  }

  // ---- Dual store C per the f32 C/D layout (C2 gets the self-loop copy).
  const int mrow = r0 + mb * 16 + ((lane & 16) ? 8 : 0);
  const int col  = c0 + (lane & 15);
  #pragma unroll
  for (int v = 0; v < 8; ++v) {
    int row = mrow + v;
    if (row < N) {
      int base = row * FDIM + col;
      C[base +  0] = acc0[v];
      C[base + 16] = acc1[v];
      C[base + 32] = acc2[v];
      C[base + 48] = acc3[v];
      C2[base +  0] = acc0[v];
      C2[base + 16] = acc1[v];
      C2[base + 32] = acc2[v];
      C2[base + 48] = acc3[v];
    }
  }
}

// ----------------------------------------------------------------------------
// SpMM scatter: Y[rows[e],:] += vals[e] * X[cols[e],:]   (64 threads per edge,
// float4 per thread). Destination (102 MB) fits in the 192 MB L2, so the
// no-return f32 atomics resolve on-chip.
// ----------------------------------------------------------------------------
__global__ __launch_bounds__(256) void spmm_scatter(const int* __restrict__ rows,
                                                    const int* __restrict__ cols,
                                                    const float* __restrict__ vals,
                                                    const float* __restrict__ X,
                                                    float* __restrict__ Y, int E) {
  int tid = blockIdx.x * blockDim.x + threadIdx.x;
  int e = tid >> 6;
  if (e >= E) return;
  int f = (tid & 63) << 2;
  int c = cols[e];
  int r = rows[e];
  float v = vals[e];
  const float4 x = *(const float4*)(X + c * FDIM + f);
  float* y = Y + r * FDIM + f;
  atomic_add_f32(y + 0, v * x.x);
  atomic_add_f32(y + 1, v * x.y);
  atomic_add_f32(y + 2, v * x.z);
  atomic_add_f32(y + 3, v * x.w);
}

// ----------------------------------------------------------------------------
// BatchNorm: per-column sum / sum-of-squares (block = 128-row stripe), then
// fold into per-column scale/bias: y = x*sc + bs.
// ----------------------------------------------------------------------------
__global__ __launch_bounds__(256) void bn_stats(const float* __restrict__ X,
                                                float* __restrict__ sums, int N) {
  int c  = threadIdx.x;            // column
  int r0 = blockIdx.x * 128;
  int rend = min(r0 + 128, N);
  float s = 0.0f, s2 = 0.0f;
  for (int r = r0; r < rend; ++r) {
    float v = X[r * FDIM + c];
    s += v;
    s2 += v * v;
  }
  atomic_add_f32(&sums[c], s);
  atomic_add_f32(&sums[FDIM + c], s2);
}

__global__ void bn_finalize(const float* __restrict__ sums, float* __restrict__ sb,
                            const float* __restrict__ gamma, const float* __restrict__ beta,
                            int N) {
  int c = threadIdx.x;
  float inv_n = 1.0f / (float)N;
  float mean = sums[c] * inv_n;
  float var  = sums[FDIM + c] * inv_n - mean * mean;
  float sc   = gamma[c] * rsqrtf(var + BN_EPS);
  sb[c]        = sc;
  sb[FDIM + c] = beta[c] - mean * sc;
}

__global__ void zero_f32(float* __restrict__ p, int n) {
  int i = blockIdx.x * blockDim.x + threadIdx.x;
  if (i < n) p[i] = 0.0f;
}

// ----------------------------------------------------------------------------
// Orchestration (copies & BN-apply fused into the GEMMs):
//   support = feature @ W1, dual-store -> buf0 (X) and buf1 (agg init)
//   agg1   += scatter(support)                 (SpMM)           -> buf1
//   sb      = bn stats over agg1               (stats+finalize)
//   Y       = relu(bn(agg1)) @ W2, dual-store  -> buf0 (X) and d_out (init)
//   out    += scatter(Y)                       (SpMM)           -> d_out
// ----------------------------------------------------------------------------
extern "C" void kernel_launch(void* const* d_in, const int* in_sizes, int n_in,
                              void* d_out, int out_size, void* d_ws, size_t ws_size,
                              hipStream_t stream) {
  const float* feature = (const float*)d_in[0];
  const int*   rows    = (const int*)d_in[1];
  const int*   cols    = (const int*)d_in[2];
  const float* vals    = (const float*)d_in[3];
  const float* W1      = (const float*)d_in[4];
  const float* W2      = (const float*)d_in[5];
  const float* gamma   = (const float*)d_in[6];
  const float* beta    = (const float*)d_in[7];
  float* out = (float*)d_out;

  const int N  = in_sizes[0] / FDIM;   // 100000
  const int E  = in_sizes[1];          // 3200000
  const int NF = N * FDIM;

  char*  ws   = (char*)d_ws;
  float* buf0 = (float*)ws;                                    // N*F f32
  float* buf1 = (float*)(ws + (size_t)NF * sizeof(float));     // N*F f32
  float* sums = (float*)(ws + 2 * (size_t)NF * sizeof(float)); // 512 f32 (32B aligned)
  float* sb   = sums + 512;                                    // 512 f32 (32B aligned)

  dim3 gemmGrid((N + 127) / 128, 4);
  const int spmmBlocks = (int)(((long long)E * 64 + 255) / 256);

  // Layer 1 GEMM: support -> buf0, self-loop init -> buf1
  gemm_wmma<false><<<gemmGrid, 256, 0, stream>>>(feature, W1, nullptr, buf0, buf1, N);
  // agg1 += (A) @ support
  spmm_scatter<<<spmmBlocks, 256, 0, stream>>>(rows, cols, vals, buf0, buf1, E);
  // BatchNorm stats -> per-column scale/bias
  zero_f32<<<2, 256, 0, stream>>>(sums, 512);
  bn_stats<<<(N + 127) / 128, 256, 0, stream>>>(buf1, sums, N);
  bn_finalize<<<1, 256, 0, stream>>>(sums, sb, gamma, beta, N);
  // Layer 2 GEMM (BN+ReLU fused into A staging): Y -> buf0, self-loop -> d_out
  gemm_wmma<true><<<gemmGrid, 256, 0, stream>>>(buf1, W2, sb, buf0, out, N);
  // out += (A) @ Y
  spmm_scatter<<<spmmBlocks, 256, 0, stream>>>(rows, cols, vals, buf0, out, E);
}
